// BatchGRUNet_33303176413479
// MI455X (gfx1250) — matrix operand order repaired and checked by reading, not compiled, not run on hardware
//
#include <hip/hip_runtime.h>
#include <math.h>

// ---------------------------------------------------------------------------
// BatchGRUNet on MI455X (gfx1250): bf16 WMMA GEMMs + f32 gate math.
//   H=300 (pad K->320), 3H=900 (pad gates->912), B=2048, L=64, N=98273.
// Round 2: preload full B tile per N-tile (20 overlapped b128 loads, one
// wait), then run the 10-WMMA chain back-to-back; prefetch next tile's W.
// ---------------------------------------------------------------------------

typedef __bf16 bf16_t;
typedef __attribute__((ext_vector_type(16))) __bf16 v16bf;
typedef __attribute__((ext_vector_type(8)))  __bf16 v8bf;
typedef __attribute__((ext_vector_type(8)))  float  v8f;

#define HH    300
#define HP    320          // padded K (10 x K=32 wmma steps)
#define G3    900
#define G3P   912          // padded gate dim (57 x 16 N-tiles)
#define BATCH 2048
#define LMAX  64
#define NTIL  57
#define KST   10
#define ROWS  (BATCH * LMAX)   // 131072

static __device__ __forceinline__ v16bf cat8(v8bf lo, v8bf hi) {
  return __builtin_shufflevector(lo, hi, 0,1,2,3,4,5,6,7,8,9,10,11,12,13,14,15);
}

// ---------------------------------------------------------------------------
// Kernel 1: packed bf16 message [B,L,320], f32 hpool [B,320], head output row.
// ---------------------------------------------------------------------------
__global__ __launch_bounds__(256) void prep_kernel(
    const float* __restrict__ node, const float* __restrict__ bias,
    const int* __restrict__ starts, const int* __restrict__ sizes,
    bf16_t* __restrict__ msgpad, float* __restrict__ hpool,
    float* __restrict__ out) {
  const int b = blockIdx.x, tid = threadIdx.x;
  const int s0 = starts[b], sz = sizes[b];

  // per-graph max-pool of raw node -> h0 (zero in the K-pad region)
  for (int k = tid; k < HP; k += 256) {
    float m = 0.0f;
    if (k < HH) {
      m = -INFINITY;
      for (int t = 0; t < sz; ++t)
        m = fmaxf(m, node[(long)(s0 + t) * HH + k]);
    }
    hpool[(long)b * HP + k] = m;
  }
  // zero-padded relu(node + bias) in bf16, K padded to 320
  for (int i = tid; i < LMAX * HP; i += 256) {
    const int t = i / HP, k = i % HP;
    float v = 0.0f;
    if (t < sz && k < HH) {
      const float x = node[(long)(s0 + t) * HH + k] + bias[k];
      v = x > 0.0f ? x : 0.0f;
    }
    msgpad[((long)b * LMAX + t) * HP + k] = (bf16_t)v;
  }
  // head row: concat(message[0], message[0])
  if (b == 0 && tid < HH) {
    const float x = node[tid] + bias[tid];
    const float v = x > 0.0f ? x : 0.0f;
    out[tid] = v;
    out[HH + tid] = v;
  }
}

// ---------------------------------------------------------------------------
// Kernel 2: f32 [900,300] weights -> bf16 [912,320] zero-padded, row major.
//   slot 0: W_ih_f  1: W_hh_f  2: W_ih_b  3: W_hh_b
// ---------------------------------------------------------------------------
__global__ __launch_bounds__(256) void wpad_kernel(
    const float* __restrict__ w0, const float* __restrict__ w1,
    const float* __restrict__ w2, const float* __restrict__ w3,
    bf16_t* __restrict__ wpad) {
  const int r = blockIdx.x;   // 0..911
  const int m = blockIdx.y;   // 0..3
  const float* src = (m == 0) ? w0 : (m == 1) ? w1 : (m == 2) ? w2 : w3;
  bf16_t* dst = wpad + ((long)m * G3P + r) * HP;
  for (int k = threadIdx.x; k < HP; k += 256) {
    const float v = (r < G3 && k < HH) ? src[(long)r * HH + k] : 0.0f;
    dst[k] = (bf16_t)v;
  }
}

// ---------------------------------------------------------------------------
// B-tile loader: 16 contiguous halves per lane per K-step, column = lane&15,
// K chunk = (lane<16 ? 0 : 16).  wrow must already include the lane offsets.
// ---------------------------------------------------------------------------
static __device__ __forceinline__ void load_btile(const bf16_t* wrow,
                                                  v16bf* btile) {
#pragma unroll
  for (int ks = 0; ks < KST; ++ks) {
    const bf16_t* p = wrow + ks * 32;
    btile[ks] = cat8(*(const v8bf*)p, *(const v8bf*)(p + 8));
  }
}

// ---------------------------------------------------------------------------
// Kernel 3: xp[dir] = msgpad @ W_ih^T + b_ih   ([131072,320]x[320,912] bf16
// WMMA, f32 accum).  8 waves/WG, one 16-row M-tile per wave, A cached in regs,
// full B tile preloaded before each WMMA chain.
// ---------------------------------------------------------------------------
__global__ __launch_bounds__(256) void xp_kernel(
    const bf16_t* __restrict__ msgpad, const bf16_t* __restrict__ wpad,
    const float* __restrict__ b_ih_f, const float* __restrict__ b_ih_b,
    float* __restrict__ xp) {
  const int dir   = blockIdx.y;
  const int lane  = threadIdx.x & 31;
  const int wave  = threadIdx.x >> 5;
  const int mbase = blockIdx.x * 128 + wave * 16;

  const bf16_t* W   = wpad + (long)(dir ? 2 : 0) * G3P * HP;
  const float*  bih = dir ? b_ih_b : b_ih_f;
  float* xpd = xp + (long)dir * ROWS * G3P;

  // A fragments (16-bit 16x32): lane row = lane&15; 16 contiguous halves at
  // k = 32*ks + base, base = (lane<16 ? 0 : 16).
  const int  am    = lane & 15;
  const int  abase = (lane < 16) ? 0 : 16;
  const bf16_t* arow = msgpad + (long)(mbase + am) * HP;
  v16bf afrag[KST];
#pragma unroll
  for (int ks = 0; ks < KST; ++ks) {
    const bf16_t* p = arow + ks * 32 + abase;
    afrag[ks] = cat8(*(const v8bf*)p, *(const v8bf*)(p + 8));
  }

  const int n     = lane & 15;           // B column = gate within tile
  const int bbase = (lane < 16) ? 0 : 16;
  const int doff  = (lane < 16) ? 0 : 8;

  for (int nt = 0; nt < NTIL; ++nt) {
    const int g = nt * 16 + n;
    const float bv = (g < G3) ? bih[g] : 0.0f;
    v8f acc = {bv, bv, bv, bv, bv, bv, bv, bv};   // bias folded into C

    // preload the entire B tile (20 overlapped b128 loads), then WMMA chain
    v16bf btile[KST];
    load_btile(W + (long)g * HP + bbase, btile);
    if (nt + 1 < NTIL)                              // warm next tile in cache
      __builtin_prefetch(W + (long)(g + 16) * HP + bbase, 0, 3);
#pragma unroll
    for (int ks = 0; ks < KST; ++ks)
      acc = __builtin_amdgcn_wmma_f32_16x16x32_bf16(
          false, afrag[ks], false, btile[ks], (short)0, acc, false, false);

    float* orow = xpd + ((long)mbase + doff) * G3P + g;
#pragma unroll
    for (int j = 0; j < 8; ++j) orow[(long)j * G3P] = acc[j];
  }
}

// ---------------------------------------------------------------------------
// Kernel 4: persistent per-batch-tile GRU scan.  16 graphs/WG, 64 steps,
// h in LDS (bf16 for WMMA, f32 for the elementwise path).
// ---------------------------------------------------------------------------
__global__ __launch_bounds__(256) void recur_kernel(
    const bf16_t* __restrict__ wpad,
    const float* __restrict__ b_hh_f, const float* __restrict__ b_hh_b,
    const float* __restrict__ xp, const float* __restrict__ hpool,
    const int* __restrict__ starts, const int* __restrict__ sizes,
    float* __restrict__ out) {
  __shared__ float gh[16][G3P];                       // 58368 B
  __shared__ float hf[16][HH];                        // 19200 B
  __shared__ __align__(16) bf16_t hb[16][HP];         // 10240 B

  const int dir  = blockIdx.y;
  const int b0   = blockIdx.x * 16;
  const int tid  = threadIdx.x;
  const int lane = tid & 31;
  const int wave = tid >> 5;

  const bf16_t* W   = wpad + (long)(dir ? 3 : 1) * G3P * HP;
  const float*  bhh = dir ? b_hh_b : b_hh_f;
  const float*  xpd = xp + (long)dir * ROWS * G3P;

  // h0 = hpool (K-pad region is zero)
  for (int i = tid; i < 16 * HP; i += 256) {
    const int m = i / HP, k = i % HP;
    const float v = hpool[(long)(b0 + m) * HP + k];
    hb[m][k] = (bf16_t)v;
    if (k < HH) hf[m][k] = v;
  }
  __syncthreads();

  const int am    = lane & 15;
  const int abase = (lane < 16) ? 0 : 16;
  const int n     = lane & 15;
  const int bbase = (lane < 16) ? 0 : 16;
  const int doff  = (lane < 16) ? 0 : 8;

  for (int t = 0; t < LMAX; ++t) {
    const int tt = dir ? (LMAX - 1 - t) : t;

    // A fragments for this step's h (shared by all N-tiles of this wave)
    v16bf afrag[KST];
#pragma unroll
    for (int ks = 0; ks < KST; ++ks) {
      const bf16_t* p = &hb[am][ks * 32 + abase];
      afrag[ks] = cat8(*(const v8bf*)p, *(const v8bf*)(p + 8));
    }

    // gh = h @ W_hh^T : 57 N-tiles split over 8 waves (wave-uniform loop)
    for (int nt = wave; nt < NTIL; nt += 8) {
      const int g = nt * 16 + n;
      v8f acc = {};

      v16bf btile[KST];
      load_btile(W + (long)g * HP + bbase, btile);
      if (nt + 8 < NTIL)                            // warm this wave's next tile
        __builtin_prefetch(W + (long)(g + 128) * HP + bbase, 0, 3);
#pragma unroll
      for (int ks = 0; ks < KST; ++ks)
        acc = __builtin_amdgcn_wmma_f32_16x16x32_bf16(
            false, afrag[ks], false, btile[ks], (short)0, acc, false, false);

#pragma unroll
      for (int j = 0; j < 8; ++j) gh[doff + j][g] = acc[j];
    }
    __syncthreads();

    // fused gate math + output scatter (f32)
    for (int i = tid; i < 16 * HH; i += 256) {
      const int m = i / HH, j = i % HH;
      const long xb = ((long)(b0 + m) * LMAX + tt) * G3P;
      const float i_r = xpd[xb + j];
      const float i_z = xpd[xb + HH + j];
      const float i_n = xpd[xb + 2 * HH + j];
      const float h_r = gh[m][j]          + bhh[j];
      const float h_z = gh[m][HH + j]     + bhh[HH + j];
      const float h_n = gh[m][2 * HH + j] + bhh[2 * HH + j];
      const float r  = 1.0f / (1.0f + expf(-(i_r + h_r)));
      const float z  = 1.0f / (1.0f + expf(-(i_z + h_z)));
      const float nn = tanhf(i_n + r * h_n);
      const float hnew = (1.0f - z) * nn + z * hf[m][j];
      hf[m][j] = hnew;
      hb[m][j] = (bf16_t)hnew;
      if (tt < sizes[b0 + m]) {
        const long row = 1 + (long)starts[b0 + m] + tt;
        out[row * (2 * HH) + (long)dir * HH + j] = hnew;
      }
    }
    __syncthreads();
  }
}

// ---------------------------------------------------------------------------
extern "C" void kernel_launch(void* const* d_in, const int* in_sizes, int n_in,
                              void* d_out, int out_size, void* d_ws, size_t ws_size,
                              hipStream_t stream) {
  const float* node   = (const float*)d_in[0];
  const float* bias   = (const float*)d_in[1];
  const float* W_ih_f = (const float*)d_in[2];
  const float* W_hh_f = (const float*)d_in[3];
  const float* b_ih_f = (const float*)d_in[4];
  const float* b_hh_f = (const float*)d_in[5];
  const float* W_ih_b = (const float*)d_in[6];
  const float* W_hh_b = (const float*)d_in[7];
  const float* b_ih_b = (const float*)d_in[8];
  const float* b_hh_b = (const float*)d_in[9];
  const int*   starts = (const int*)d_in[10];
  const int*   sizes  = (const int*)d_in[11];
  float* out = (float*)d_out;

  // workspace layout (256B aligned sections)
  char* ws = (char*)d_ws;
  bf16_t* wpad   = (bf16_t*)(ws);                                  //  2,334,720 B
  bf16_t* msgpad = (bf16_t*)(ws + 2334720);                        // 83,886,080 B
  float*  hpool  = (float*) (ws + 2334720 + 83886080);             //  2,621,440 B
  float*  xp     = (float*) (ws + 2334720 + 83886080 + 2621440);   // 956,301,312 B

  wpad_kernel<<<dim3(G3P, 4), 256, 0, stream>>>(
      W_ih_f, W_hh_f, W_ih_b, W_hh_b, wpad);
  prep_kernel<<<dim3(BATCH), 256, 0, stream>>>(
      node, bias, starts, sizes, msgpad, hpool, out);
  xp_kernel<<<dim3(ROWS / 128, 2), 256, 0, stream>>>(
      msgpad, wpad, b_ih_f, b_ih_b, xp);
  recur_kernel<<<dim3(BATCH / 16, 2), 256, 0, stream>>>(
      wpad, b_hh_f, b_hh_b, xp, hpool, starts, sizes, out);
}
